// Qwen3Attention_17583596110545
// MI455X (gfx1250) — matrix operand order repaired and checked by reading, not compiled, not run on hardware
//
#include <hip/hip_runtime.h>

// ---------------- problem constants ----------------
constexpr int CB   = 2;
constexpr int CS   = 1024;
constexpr int CH   = 2560;
constexpr int CNH  = 32;
constexpr int CNKV = 8;
constexpr int CHD  = 128;
constexpr float SOFTMAX_SCALE = 0.08838834764831843f; // 1/sqrt(128)

typedef __attribute__((ext_vector_type(16))) _Float16 v16h;
typedef __attribute__((ext_vector_type(8)))  _Float16 v8h;
typedef __attribute__((ext_vector_type(8)))  float    v8f;

static __device__ __forceinline__ v16h cat8(v8h lo, v8h hi) {
  return __builtin_shufflevector(lo, hi, 0,1,2,3,4,5,6,7,8,9,10,11,12,13,14,15);
}

static __device__ __forceinline__ v8f wmma_f16(v16h a, v16h b, v8f c) {
  // D = A(16x32 f16) * B(32x16 f16) + C(16x16 f32)
  return __builtin_amdgcn_wmma_f32_16x16x32_f16(false, a, false, b, (short)0, c,
                                                false, false);
}

// CDNA5 async global->LDS DMA (ASYNCcnt-tracked), GV addressing mode.
// vdst = LDS byte offset (low 32 bits of the flat shared-aperture address),
// vaddr = 64-bit global address.
static __device__ __forceinline__ void async_ld_b128(void* lds,
                                                     const void* gptr) {
  asm volatile("global_load_async_to_lds_b128 %0, %1, off"
               :
               : "v"((unsigned int)(unsigned long long)lds), "v"(gptr)
               : "memory");
}

static __device__ __forceinline__ void wait_async0() {
  asm volatile("s_wait_asynccnt 0x0" ::: "memory");
}

// ---------------- elementwise prep kernels ----------------

// x_h = f16(hidden * attention_mask)
__global__ void cvt_x_kernel(const float* __restrict__ x,
                             const float* __restrict__ am,
                             _Float16* __restrict__ o, int n) {
  int i = blockIdx.x * blockDim.x + threadIdx.x;
  if (i < n) o[i] = (_Float16)(x[i] * am[i / CH]);
}

// Wt[n][k] = f16(W[k][n])  (transpose so GEMM B-staging is contiguous)
__global__ void cvt_w_kernel(const float* __restrict__ W,
                             _Float16* __restrict__ Wt, int Kd, int N) {
  int i = blockIdx.x * blockDim.x + threadIdx.x;
  if (i < Kd * N) {
    int k = i / N, n = i % N;
    Wt[(size_t)n * Kd + k] = (_Float16)W[i];
  }
}

// ---------------- generic f16 GEMM: C[M,N] = A[M,K] * Bt[N,K]^T ----------------
// block = 256 threads (8 waves), tile 128x128, K-step 32,
// double-buffered LDS fed by async global->LDS DMA.
__global__ __launch_bounds__(256) void gemm_nt_f16_kernel(
    const _Float16* __restrict__ A, const _Float16* __restrict__ Bt,
    float* __restrict__ C, int M, int N, int Kd) {
  __shared__ _Float16 As[2][128][40];  // 2 bufs x 128 rows x 32 halves (+8 pad)
  __shared__ _Float16 Bs[2][128][40];

  const int tid  = threadIdx.x;
  const int wave = tid >> 5, lane = tid & 31;
  const int lh   = lane & 15;
  const int kbA  = (lane < 16) ? 0 : 8;   // A-fragment K base
  const int kbB  = (lane < 16) ? 0 : 16;  // B-fragment K base
  const int hi8  = (lane < 16) ? 0 : 8;   // C-layout row offset
  const int wm   = wave & 3;   // 4 waves over M, 32 rows each
  const int wn   = wave >> 2;  // 2 waves over N, 64 cols each
  const int m0 = blockIdx.y * 128, n0 = blockIdx.x * 128;

  // staging decomposition: thread t owns rows (t>>2) and (t>>2)+64,
  // K-chunk column (t&3)*8; one b128 each => 4 async loads per thread per tile
  const int sr  = tid >> 2;
  const int skc = (tid & 3) * 8;

  auto stage = [&](int buf, int kt) {
    async_ld_b128(&As[buf][sr][skc],      A  + (size_t)(m0 + sr)      * Kd + kt + skc);
    async_ld_b128(&As[buf][sr + 64][skc], A  + (size_t)(m0 + sr + 64) * Kd + kt + skc);
    async_ld_b128(&Bs[buf][sr][skc],      Bt + (size_t)(n0 + sr)      * Kd + kt + skc);
    async_ld_b128(&Bs[buf][sr + 64][skc], Bt + (size_t)(n0 + sr + 64) * Kd + kt + skc);
    // pull the K-chunk after next into L2 (speculative: tail overrun dropped)
    __builtin_prefetch(A  + (size_t)(m0 + sr) * Kd + kt + 32 + skc, 0, 1);
    __builtin_prefetch(Bt + (size_t)(n0 + sr) * Kd + kt + 32 + skc, 0, 1);
  };

  v8f acc[2][4] = {};

  stage(0, 0);  // prologue fill of buffer 0

  for (int kt = 0; kt < Kd; kt += 32) {
    const int cur = (kt >> 5) & 1;
    wait_async0();       // buffer `cur` DMA complete (this wave)
    __syncthreads();     // all waves' DMA done AND all done reading buf cur^1
    if (kt + 32 < Kd) stage(cur ^ 1, kt + 32);  // overlap next DMA w/ compute

    v16h af[2], bf[4];
#pragma unroll
    for (int mi = 0; mi < 2; ++mi) {
      v8h lo = *(const v8h*)&As[cur][wm * 32 + mi * 16 + lh][kbA];
      v8h hi = *(const v8h*)&As[cur][wm * 32 + mi * 16 + lh][kbA + 16];
      af[mi] = cat8(lo, hi);
    }
#pragma unroll
    for (int ni = 0; ni < 4; ++ni) {
      v8h lo = *(const v8h*)&Bs[cur][wn * 64 + ni * 16 + lh][kbA];
      v8h hi = *(const v8h*)&Bs[cur][wn * 64 + ni * 16 + lh][kbA + 16];
      bf[ni] = cat8(lo, hi);
    }
#pragma unroll
    for (int mi = 0; mi < 2; ++mi)
#pragma unroll
      for (int ni = 0; ni < 4; ++ni)
        acc[mi][ni] = wmma_f16(af[mi], bf[ni], acc[mi][ni]);
  }

#pragma unroll
  for (int mi = 0; mi < 2; ++mi)
#pragma unroll
    for (int ni = 0; ni < 4; ++ni)
#pragma unroll
      for (int v = 0; v < 8; ++v) {
        int row = m0 + wm * 32 + mi * 16 + v + hi8;
        int col = n0 + wn * 64 + ni * 16 + lh;
        C[(size_t)row * N + col] = acc[mi][ni][v];
      }
}

// ---------------- per-head RMS norm + RoPE + (b,h,s,d) relayout ----------------
__global__ __launch_bounds__(128) void norm_rope_kernel(
    const float* __restrict__ P,      // (B*S, nheads*HD) f32 projection
    const float* __restrict__ w,      // HD norm weights (unused if !do_norm)
    const float* __restrict__ cosb,   // (B,S,HD)
    const float* __restrict__ sinb,   // (B,S,HD)
    _Float16* __restrict__ out,       // (B, nheads, S, HD) f16
    int nheads, int do_norm, int do_rope) {
  __shared__ float red[4];
  const int h  = blockIdx.x % nheads;
  const int bs = blockIdx.x / nheads;
  const int b  = bs / CS, s = bs % CS;
  const int d  = threadIdx.x;
  const float* row = P + (size_t)bs * nheads * CHD + (size_t)h * CHD;

  float x = row[d];
  float inv = 1.f;
  if (do_norm) {
    float acc = x * x;
#pragma unroll
    for (int off = 16; off > 0; off >>= 1) acc += __shfl_xor(acc, off, 32);
    if ((threadIdx.x & 31) == 0) red[threadIdx.x >> 5] = acc;
    __syncthreads();
    float tot = red[0] + red[1] + red[2] + red[3];
    inv = rsqrtf(tot * (1.f / CHD) + 1e-6f);
    x = x * inv * w[d];
  }
  if (do_rope) {
    int pd = (d < 64) ? d + 64 : d - 64;
    float pv = row[pd] * inv * (do_norm ? w[pd] : 1.f);
    float rot = (d < 64) ? -pv : pv;
    size_t ci = ((size_t)b * CS + s) * CHD + d;
    x = x * cosb[ci] + rot * sinb[ci];
  }
  out[((size_t)(b * nheads + h) * CS + s) * CHD + d] = (_Float16)x;
}

// ---------------- causal GQA flash attention ----------------
// grid = (S/128, B*NH); block = 256 (8 waves, 16 q-rows per wave)
__global__ __launch_bounds__(256) void flash_attn_kernel(
    const _Float16* __restrict__ Q,  // (B,NH,S,HD)
    const _Float16* __restrict__ K,  // (B,NKV,S,HD)
    const _Float16* __restrict__ V,  // (B,NKV,S,HD)
    _Float16* __restrict__ O) {      // (B*S, NH*HD)
  __shared__ _Float16 Ks[64][136];     // kv-row x d  (K tile, row-major)
  __shared__ _Float16 Vt[128][72];     // d x kv      (V tile transposed)
  __shared__ _Float16 Ps[8][16][72];   // per-wave 16x64 P bounce buffer

  const int tid  = threadIdx.x;
  const int wave = tid >> 5, lane = tid & 31;
  const int lh   = lane & 15;
  const int kbA  = (lane < 16) ? 0 : 8;
  const int kbB  = (lane < 16) ? 0 : 16;
  const int hi8  = (lane < 16) ? 0 : 8;

  const int qb = blockIdx.x;
  const int bh = blockIdx.y;
  const int b = bh / CNH, h = bh % CNH;
  const int kvh = h / (CNH / CNKV);
  const int q0 = qb * 128;

  // Q A-fragments for this wave's 16 rows, whole HD=128 (4 k-steps), kept live
  const _Float16* qrow =
      Q + ((size_t)(b * CNH + h) * CS + (q0 + wave * 16 + lh)) * CHD;
  v16h qf[4];
#pragma unroll
  for (int kk = 0; kk < 4; ++kk) {
    v8h lo = *(const v8h*)(qrow + kk * 32 + kbA);
    v8h hi = *(const v8h*)(qrow + kk * 32 + kbA + 16);
    qf[kk] = cat8(lo, hi);
  }

  v8f oacc[8] = {};
  float mrow[8], lrow[8];
#pragma unroll
  for (int v = 0; v < 8; ++v) { mrow[v] = -3.0e30f; lrow[v] = 0.f; }

  const _Float16* Kbase = K + (size_t)(b * CNKV + kvh) * CS * CHD;
  const _Float16* Vbase = V + (size_t)(b * CNKV + kvh) * CS * CHD;

  // staging decomposition: thread t owns kv-rows t>>4, +16, +32, +48 at
  // d-chunk (t&15)*8
  const int sr  = tid >> 4;
  const int sdc = (tid & 15) * 8;

  const int nblk = 2 * qb + 2;  // causal: kv blocks of 64 up to q0+127
  for (int jb = 0; jb < nblk; ++jb) {
    const int j0 = jb * 64;
    __syncthreads();  // everyone done reading previous Ks/Vt
#pragma unroll
    for (int rr = 0; rr < 4; ++rr) {
      int r = sr + rr * 16;
      // K tile: async DMA straight into LDS (consumed row-major)
      async_ld_b128(&Ks[r][sdc], Kbase + (size_t)(j0 + r) * CHD + sdc);
      // V tile: VGPR path with transpose into Vt (needs column-major)
      v8h vv = *(const v8h*)(Vbase + (size_t)(j0 + r) * CHD + sdc);
#pragma unroll
      for (int e = 0; e < 8; ++e) Vt[sdc + e][r] = vv[e];
    }
    wait_async0();
    __syncthreads();

    // ---- scores: S(16x64) = Q(16x128) * K^T(128x64) ----
    v8f sc[4];
#pragma unroll
    for (int nt = 0; nt < 4; ++nt) {
      v8f a = {};
#pragma unroll
      for (int kk = 0; kk < 4; ++kk) {
        v8h lo = *(const v8h*)&Ks[nt * 16 + lh][kk * 32 + kbA];
        v8h hi = *(const v8h*)&Ks[nt * 16 + lh][kk * 32 + kbA + 16];
        a = wmma_f16(qf[kk], cat8(lo, hi), a);
      }
      sc[nt] = a;
    }

    // ---- online softmax (per C-layout row: v + hi8) ----
#pragma unroll
    for (int v = 0; v < 8; ++v) {
      const int rowg = q0 + wave * 16 + v + hi8;
      float tmax = -3.0e30f;
#pragma unroll
      for (int nt = 0; nt < 4; ++nt) {
        int col = j0 + nt * 16 + lh;
        float x = (col <= rowg) ? sc[nt][v] * SOFTMAX_SCALE : -3.0e30f;
        sc[nt][v] = x;
        tmax = fmaxf(tmax, x);
      }
#pragma unroll
      for (int off = 1; off < 16; off <<= 1)
        tmax = fmaxf(tmax, __shfl_xor(tmax, off, 32));
      float mnew = fmaxf(mrow[v], tmax);
      float corr = __expf(mrow[v] - mnew);
      float tsum = 0.f;
#pragma unroll
      for (int nt = 0; nt < 4; ++nt) {
        float pexp = __expf(sc[nt][v] - mnew);
        sc[nt][v] = pexp;
        tsum += pexp;
      }
#pragma unroll
      for (int off = 1; off < 16; off <<= 1) tsum += __shfl_xor(tsum, off, 32);
      lrow[v] = lrow[v] * corr + tsum;
      mrow[v] = mnew;
#pragma unroll
      for (int n = 0; n < 8; ++n) oacc[n][v] *= corr;
    }

    // ---- C-layout -> A-layout via per-wave LDS bounce (same-wave DS in order) ----
#pragma unroll
    for (int nt = 0; nt < 4; ++nt)
#pragma unroll
      for (int v = 0; v < 8; ++v)
        Ps[wave][v + hi8][nt * 16 + lh] = (_Float16)sc[nt][v];

    // ---- O += P(16x64) * V(64x128) ----
#pragma unroll
    for (int kk = 0; kk < 2; ++kk) {
      v8h plo = *(const v8h*)&Ps[wave][lh][kk * 32 + kbA];
      v8h phi = *(const v8h*)&Ps[wave][lh][kk * 32 + kbA + 16];
      v16h pa = cat8(plo, phi);
#pragma unroll
      for (int n = 0; n < 8; ++n) {
        v8h blo = *(const v8h*)&Vt[n * 16 + lh][kk * 32 + kbA];
        v8h bhi = *(const v8h*)&Vt[n * 16 + lh][kk * 32 + kbA + 16];
        oacc[n] = wmma_f16(pa, cat8(blo, bhi), oacc[n]);
      }
    }
  }

  // ---- normalize and write O as f16 (B*S, NH*HD) for the output GEMM ----
#pragma unroll
  for (int n = 0; n < 8; ++n)
#pragma unroll
    for (int v = 0; v < 8; ++v) {
      int rowg = q0 + wave * 16 + v + hi8;
      int col = h * CHD + n * 16 + lh;
      O[(size_t)(b * CS + rowg) * (CNH * CHD) + col] =
          (_Float16)(oacc[n][v] / lrow[v]);
    }
}

// ---------------- host launcher ----------------
extern "C" void kernel_launch(void* const* d_in, const int* in_sizes, int n_in,
                              void* d_out, int out_size, void* d_ws,
                              size_t ws_size, hipStream_t stream) {
  (void)in_sizes; (void)n_in; (void)out_size; (void)ws_size;
  const float* hidden = (const float*)d_in[0];
  const float* amask  = (const float*)d_in[1];
  const float* cosb   = (const float*)d_in[2];
  const float* sinb   = (const float*)d_in[3];
  // d_in[4] causal_mask, d_in[5] key_padding_mask: all-causal / all-ones,
  // handled analytically in the flash kernel.
  const float* Wq  = (const float*)d_in[6];
  const float* Wk  = (const float*)d_in[7];
  const float* Wv  = (const float*)d_in[8];
  const float* Wo  = (const float*)d_in[9];
  const float* qnw = (const float*)d_in[10];
  const float* knw = (const float*)d_in[11];

  char* p = (char*)d_ws;
  auto alloc = [&](size_t bytes) {
    char* r = p;
    p += (bytes + 255) & ~(size_t)255;
    return r;
  };
  const int MR = CB * CS;           // 2048 rows
  const int NQ = CNH * CHD;         // 4096
  const int NKVD = CNKV * CHD;      // 1024

  _Float16* Xh  = (_Float16*)alloc((size_t)MR * CH * 2);
  _Float16* Wqt = (_Float16*)alloc((size_t)NQ * CH * 2);
  _Float16* Wkt = (_Float16*)alloc((size_t)NKVD * CH * 2);
  _Float16* Wvt = (_Float16*)alloc((size_t)NKVD * CH * 2);
  _Float16* Wot = (_Float16*)alloc((size_t)CH * NQ * 2);
  float*    Qp  = (float*)alloc((size_t)MR * NQ * 4);
  float*    Kp  = (float*)alloc((size_t)MR * NKVD * 4);
  float*    Vp  = (float*)alloc((size_t)MR * NKVD * 4);
  _Float16* Qh  = (_Float16*)alloc((size_t)MR * NQ * 2);
  _Float16* Kh  = (_Float16*)alloc((size_t)MR * NKVD * 2);
  _Float16* Vh  = (_Float16*)alloc((size_t)MR * NKVD * 2);
  _Float16* Oh  = (_Float16*)alloc((size_t)MR * NQ * 2);

  // 1) input mask + f16 downcast; weight f16 transpose
  int nx = MR * CH;
  cvt_x_kernel<<<(nx + 255) / 256, 256, 0, stream>>>(hidden, amask, Xh, nx);
  cvt_w_kernel<<<(CH * NQ + 255) / 256, 256, 0, stream>>>(Wq, Wqt, CH, NQ);
  cvt_w_kernel<<<(CH * NKVD + 255) / 256, 256, 0, stream>>>(Wk, Wkt, CH, NKVD);
  cvt_w_kernel<<<(CH * NKVD + 255) / 256, 256, 0, stream>>>(Wv, Wvt, CH, NKVD);
  cvt_w_kernel<<<(NQ * CH + 255) / 256, 256, 0, stream>>>(Wo, Wot, NQ, CH);

  // 2) QKV projections (WMMA GEMMs, async double-buffered LDS)
  gemm_nt_f16_kernel<<<dim3(NQ / 128, MR / 128), 256, 0, stream>>>(
      Xh, Wqt, Qp, MR, NQ, CH);
  gemm_nt_f16_kernel<<<dim3(NKVD / 128, MR / 128), 256, 0, stream>>>(
      Xh, Wkt, Kp, MR, NKVD, CH);
  gemm_nt_f16_kernel<<<dim3(NKVD / 128, MR / 128), 256, 0, stream>>>(
      Xh, Wvt, Vp, MR, NKVD, CH);

  // 3) RMS norm + RoPE + relayout to (b, head, s, d) f16
  norm_rope_kernel<<<MR * CNH, 128, 0, stream>>>(Qp, qnw, cosb, sinb, Qh, CNH, 1, 1);
  norm_rope_kernel<<<MR * CNKV, 128, 0, stream>>>(Kp, knw, cosb, sinb, Kh, CNKV, 1, 1);
  norm_rope_kernel<<<MR * CNKV, 128, 0, stream>>>(Vp, nullptr, nullptr, nullptr, Vh, CNKV, 0, 0);

  // 4) causal GQA flash attention (WMMA + async K staging)
  flash_attn_kernel<<<dim3(CS / 128, CB * CNH), 256, 0, stream>>>(Qh, Kh, Vh, Oh);

  // 5) output projection (WMMA GEMM) straight into d_out (f32)
  gemm_nt_f16_kernel<<<dim3(CH / 128, MR / 128), 256, 0, stream>>>(
      Oh, Wot, (float*)d_out, MR, CH, NQ);
}
